// GNNEncoder_15496242004331
// MI455X (gfx1250) — matrix operand order repaired and checked by reading, not compiled, not run on hardware
//
#include <hip/hip_runtime.h>

#define EPSV 1e-5f
#define B_   32
#define N_   32
#define C_   512
#define H_   8
#define HD_  64
#define L2_  (N_ * N_)

typedef __attribute__((ext_vector_type(16))) __bf16 v16bf;
typedef __attribute__((ext_vector_type(8)))  __bf16 v8bf;
typedef __attribute__((ext_vector_type(8)))  float  v8f;

__device__ __forceinline__ unsigned short f2bf_u(float f) {
  union { float f; unsigned u; } x; x.f = f;
  unsigned r = x.u + 0x7FFFu + ((x.u >> 16) & 1u);
  return (unsigned short)(r >> 16);
}
__device__ __forceinline__ __bf16 f2bf(float f) {
  union { unsigned short u; __bf16 b; } r; r.u = f2bf_u(f); return r.b;
}

// CDNA5 16-bit WMMA fragment loads (CDNA5 ISA layouts) — contiguous b128 pairs
__device__ __forceinline__ v16bf load_afrag(const __bf16* Ar, int kb, int half) {
  union { v16bf v; v8bf h[2]; } a;
  a.h[0] = *(const v8bf*)(Ar + kb + half * 8);        // K = kb+half*8 .. +7
  a.h[1] = *(const v8bf*)(Ar + kb + 16 + half * 8);   // K = kb+16+half*8 .. +7
  return a.v;
}
__device__ __forceinline__ v16bf load_bfrag(const __bf16* Wr, int kb, int half) {
  union { v16bf v; v8bf h[2]; } b;
  b.h[0] = *(const v8bf*)(Wr + kb + half * 8);        // K = kb+half*8 .. +7
  b.h[1] = *(const v8bf*)(Wr + kb + 16 + half * 8);   // K = kb+16+half*8 .. +7
  return b.v;
}

// ---------------------------------------------------------------------------
// GEMM: C[M,N] = act(A[M,K] @ W[N,K]^T + bias); bf16 in, f32 accum.
// 32x64 output slab per wave32: 8 WMMAs per K-step over 12 b128 loads.
// ---------------------------------------------------------------------------
__global__ __launch_bounds__(256) void gemm_nt_kernel(
    const __bf16* __restrict__ A, const __bf16* __restrict__ W,
    const float* __restrict__ bias, float* __restrict__ Cf, __bf16* __restrict__ Cb,
    int M, int N, int K, int lda, int ldw, int ldc, int relu)
{
  int lane = threadIdx.x & 31, wid = threadIdx.x >> 5;
  int tilesN = N >> 6;
  long long tile = (long long)blockIdx.x * 8 + wid;
  long long nt = (long long)(M >> 5) * tilesN;
  if (tile >= nt) return;
  int tm = (int)(tile / tilesN), tn = (int)(tile % tilesN);
  int m0 = tm << 5, n0 = tn << 6;
  int half = lane >> 4, lm = lane & 15;
  const __bf16* Ar0 = A + (size_t)(m0 + lm) * lda;
  const __bf16* Ar1 = A + (size_t)(m0 + 16 + lm) * lda;
  const __bf16* Wp[4];
#pragma unroll
  for (int q = 0; q < 4; ++q) Wp[q] = W + (size_t)(n0 + 16*q + lm) * ldw;
  v8f acc0[4], acc1[4];
#pragma unroll
  for (int q = 0; q < 4; ++q) {
    acc0[q] = (v8f){0.f,0.f,0.f,0.f,0.f,0.f,0.f,0.f};
    acc1[q] = (v8f){0.f,0.f,0.f,0.f,0.f,0.f,0.f,0.f};
  }
  for (int kb = 0; kb < K; kb += 32) {
    if (kb + 32 < K) {                                // -> global_prefetch_b8
      __builtin_prefetch(Ar0 + kb + 32, 0, 3);
      __builtin_prefetch(Ar1 + kb + 32, 0, 3);
      __builtin_prefetch(Wp[0] + kb + 32, 0, 3);
      __builtin_prefetch(Wp[2] + kb + 32, 0, 3);
    }
    v16bf a0 = load_afrag(Ar0, kb, half);
    v16bf a1 = load_afrag(Ar1, kb, half);
#pragma unroll
    for (int q = 0; q < 4; ++q) {
      v16bf bq = load_bfrag(Wp[q], kb, half);
      acc0[q] = __builtin_amdgcn_wmma_f32_16x16x32_bf16(false, a0, false, bq, (short)0, acc0[q], false, false);
      acc1[q] = __builtin_amdgcn_wmma_f32_16x16x32_bf16(false, a1, false, bq, (short)0, acc1[q], false, false);
    }
  }
#pragma unroll
  for (int q = 0; q < 4; ++q) {
    int col = n0 + 16*q + lm;
    float bv = bias ? bias[col] : 0.f;
#pragma unroll
    for (int j = 0; j < 8; ++j) {
      int mA = m0 + j + half * 8;
      int mB = mA + 16;
      float v0 = acc0[q][j] + bv, v1 = acc1[q][j] + bv;
      if (relu) { v0 = fmaxf(v0, 0.f); v1 = fmaxf(v1, 0.f); }
      if (Cf) { Cf[(size_t)mA * ldc + col] = v0; Cf[(size_t)mB * ldc + col] = v1; }
      if (Cb) { Cb[(size_t)mA * ldc + col] = f2bf(v0); Cb[(size_t)mB * ldc + col] = f2bf(v1); }
    }
  }
}

// ---------------------------------------------------------------------------
// Attention scores: S[bh, Lq, Lk] = scale * Q_sub @ K_sub^T   (hd = 64)
// ---------------------------------------------------------------------------
__global__ __launch_bounds__(256) void attn_scores_kernel(
    const __bf16* __restrict__ Q, int ldq, const __bf16* __restrict__ Kp, int ldk,
    float* __restrict__ S, int Lq, int Lk, float scale, int bh_base)
{
  int lane = threadIdx.x & 31, wid = threadIdx.x >> 5;
  int bh = bh_base + blockIdx.y;
  int b = bh / H_, h = bh % H_;
  int tilesN = Lk >> 5;
  long long tile = (long long)blockIdx.x * 8 + wid;
  long long nt = (long long)(Lq >> 4) * tilesN;
  if (tile >= nt) return;
  int tm = (int)(tile / tilesN), tn = (int)(tile % tilesN);
  int m0 = tm << 4, n0 = tn << 5;
  int half = lane >> 4, lm = lane & 15;
  const __bf16* Ar = Q  + (size_t)(b * Lq + m0 + lm) * ldq + h * HD_;
  const __bf16* W0 = Kp + (size_t)(b * Lk + n0 + lm) * ldk + h * HD_;
  const __bf16* W1 = Kp + (size_t)(b * Lk + n0 + 16 + lm) * ldk + h * HD_;
  v8f acc0 = {0.f,0.f,0.f,0.f,0.f,0.f,0.f,0.f};
  v8f acc1 = {0.f,0.f,0.f,0.f,0.f,0.f,0.f,0.f};
#pragma unroll
  for (int kb = 0; kb < HD_; kb += 32) {
    v16bf a  = load_afrag(Ar, kb, half);
    v16bf b0 = load_bfrag(W0, kb, half);
    v16bf b1 = load_bfrag(W1, kb, half);
    acc0 = __builtin_amdgcn_wmma_f32_16x16x32_bf16(false, a, false, b0, (short)0, acc0, false, false);
    acc1 = __builtin_amdgcn_wmma_f32_16x16x32_bf16(false, a, false, b1, (short)0, acc1, false, false);
  }
  float* Sb = S + (size_t)blockIdx.y * Lq * Lk;
#pragma unroll
  for (int j = 0; j < 8; ++j) {
    int m = m0 + j + half * 8;
    Sb[(size_t)m * Lk + n0 + lm]      = acc0[j] * scale;
    Sb[(size_t)m * Lk + n0 + 16 + lm] = acc1[j] * scale;
  }
}

// ---------------------------------------------------------------------------
// O = P @ V via NT form against pre-transposed V (VT[d, k], bf16)
// ---------------------------------------------------------------------------
__global__ __launch_bounds__(256) void attn_av_kernel(
    const __bf16* __restrict__ P, const __bf16* __restrict__ VT,
    __bf16* __restrict__ O, int ldo, int Lq, int Lk, int bh_base)
{
  int lane = threadIdx.x & 31, wid = threadIdx.x >> 5;
  int bh = bh_base + blockIdx.y;
  int b = bh / H_, h = bh % H_;
  int tilesN = HD_ >> 5;   // 2
  long long tile = (long long)blockIdx.x * 8 + wid;
  long long nt = (long long)(Lq >> 4) * tilesN;
  if (tile >= nt) return;
  int tm = (int)(tile / tilesN), tn = (int)(tile % tilesN);
  int m0 = tm << 4, n0 = tn << 5;
  int half = lane >> 4, lm = lane & 15;
  const __bf16* Ar = P + (size_t)blockIdx.y * Lq * Lk + (size_t)(m0 + lm) * Lk;
  const __bf16* Vb = VT + ((size_t)b * C_ + h * HD_) * Lk;
  const __bf16* W0 = Vb + (size_t)(n0 + lm) * Lk;
  const __bf16* W1 = Vb + (size_t)(n0 + 16 + lm) * Lk;
  v8f acc0 = {0.f,0.f,0.f,0.f,0.f,0.f,0.f,0.f};
  v8f acc1 = {0.f,0.f,0.f,0.f,0.f,0.f,0.f,0.f};
  for (int kb = 0; kb < Lk; kb += 32) {
    v16bf a  = load_afrag(Ar, kb, half);
    v16bf b0 = load_bfrag(W0, kb, half);
    v16bf b1 = load_bfrag(W1, kb, half);
    acc0 = __builtin_amdgcn_wmma_f32_16x16x32_bf16(false, a, false, b0, (short)0, acc0, false, false);
    acc1 = __builtin_amdgcn_wmma_f32_16x16x32_bf16(false, a, false, b1, (short)0, acc1, false, false);
  }
#pragma unroll
  for (int j = 0; j < 8; ++j) {
    int m = m0 + j + half * 8;
    size_t ro = (size_t)(b * Lq + m) * ldo + h * HD_;
    O[ro + n0 + lm]      = f2bf(acc0[j]);
    O[ro + n0 + 16 + lm] = f2bf(acc1[j]);
  }
}

// ---------------------------------------------------------------------------
__global__ void softmax_rows_kernel(float* __restrict__ S, __bf16* __restrict__ P, int Lk) {
  __shared__ float red[128];
  size_t base = (size_t)blockIdx.x * Lk;
  int tid = threadIdx.x;
  float mx = -3.4e38f;
  for (int i = tid; i < Lk; i += 128) mx = fmaxf(mx, S[base + i]);
  red[tid] = mx; __syncthreads();
  for (int s = 64; s > 0; s >>= 1) { if (tid < s) red[tid] = fmaxf(red[tid], red[tid+s]); __syncthreads(); }
  mx = red[0]; __syncthreads();
  float sum = 0.f;
  for (int i = tid; i < Lk; i += 128) { float e = __expf(S[base+i] - mx); S[base+i] = e; sum += e; }
  red[tid] = sum; __syncthreads();
  for (int s = 64; s > 0; s >>= 1) { if (tid < s) red[tid] += red[tid+s]; __syncthreads(); }
  float inv = 1.f / red[0];
  for (int i = tid; i < Lk; i += 128) P[base+i] = f2bf(S[base+i] * inv);
}

__global__ void layernorm_kernel(
    const float* __restrict__ X, const float* __restrict__ R,
    const float* __restrict__ g, const float* __restrict__ bt,
    const float* __restrict__ add, float* __restrict__ Yf, __bf16* __restrict__ Yb, int Cd)
{
  __shared__ float r1[256], r2[256];
  size_t base = (size_t)blockIdx.x * Cd;
  int tid = threadIdx.x;
  float s1 = 0.f, s2 = 0.f;
  for (int i = tid; i < Cd; i += 256) { float v = X[base+i] + R[base+i]; s1 += v; s2 += v*v; }
  r1[tid] = s1; r2[tid] = s2; __syncthreads();
  for (int s = 128; s > 0; s >>= 1) { if (tid < s) { r1[tid] += r1[tid+s]; r2[tid] += r2[tid+s]; } __syncthreads(); }
  float m = r1[0] / Cd, var = r2[0] / Cd - m*m, rs = rsqrtf(var + EPSV);
  for (int i = tid; i < Cd; i += 256) {
    float v = X[base+i] + R[base+i];
    float o = g[i]*(v - m)*rs + bt[i];
    if (add) o += add[base+i];
    if (Yf) Yf[base+i] = o;
    if (Yb) Yb[base+i] = f2bf(o);
  }
}

__global__ void groupnorm_relu_kernel(
    const float* __restrict__ X, const float* __restrict__ g,
    const float* __restrict__ bt, float* __restrict__ Yf, __bf16* __restrict__ Yb,
    int Ch, int G)
{
  int gs = Ch / G;
  int bi = blockIdx.x / G, gr = blockIdx.x % G;
  int lane = threadIdx.x;
  size_t base = (size_t)bi * Ch + (size_t)gr * gs;
  float s1 = 0.f, s2 = 0.f;
  for (int i = lane; i < gs; i += 32) { float v = X[base+i]; s1 += v; s2 += v*v; }
#pragma unroll
  for (int m = 16; m > 0; m >>= 1) { s1 += __shfl_xor(s1, m, 32); s2 += __shfl_xor(s2, m, 32); }
  float mn = s1 / gs, var = s2 / gs - mn*mn, rs = rsqrtf(var + EPSV);
  for (int i = lane; i < gs; i += 32) {
    int c = gr*gs + i;
    float o = fmaxf(g[c]*(X[base+i] - mn)*rs + bt[c], 0.f);
    if (Yf) Yf[base+i] = o;
    if (Yb) Yb[base+i] = f2bf(o);
  }
}

__global__ void bn_stats_kernel(const float* __restrict__ X, int Bn, int Ch, int L,
                                float* __restrict__ mean, float* __restrict__ inv)
{
  __shared__ float r1[256], r2[256];
  int ch = blockIdx.x, tid = threadIdx.x;
  float s1 = 0.f, s2 = 0.f;
  int tot = Bn * L;
  for (int t = tid; t < tot; t += 256) {
    int b = t / L, l = t - b * L;
    float v = X[((size_t)b * Ch + ch) * L + l];
    s1 += v; s2 += v*v;
  }
  r1[tid] = s1; r2[tid] = s2; __syncthreads();
  for (int s = 128; s > 0; s >>= 1) { if (tid < s) { r1[tid] += r1[tid+s]; r2[tid] += r2[tid+s]; } __syncthreads(); }
  if (tid == 0) {
    float m = r1[0] / tot, var = r2[0] / tot - m*m;
    mean[ch] = m; inv[ch] = rsqrtf(var + EPSV);
  }
}

// ------------------------------ elementwise --------------------------------
#define GS_LOOP(i, total) \
  for (size_t i = (size_t)blockIdx.x * blockDim.x + threadIdx.x; i < (total); i += (size_t)gridDim.x * blockDim.x)

__global__ void cvt_bf16_kernel(const float* __restrict__ X, __bf16* __restrict__ Y, size_t total) {
  GS_LOOP(idx, total) Y[idx] = f2bf(X[idx]);
}
__global__ void transpose_node_kernel(const float* __restrict__ node, __bf16* __restrict__ xt) {
  GS_LOOP(idx, (size_t)B_ * N_ * C_) {
    size_t b = idx / (N_ * C_), rem = idx % (N_ * C_);
    size_t k = rem / C_, c = rem % C_;
    xt[b * (size_t)(C_ * N_) + c * N_ + k] = f2bf(node[idx]);
  }
}
// VT[b, d, k] = KV[b, k, voff + d]  (bf16 -> bf16), d in [0, C_)
__global__ void transpose_v_kernel(const __bf16* __restrict__ KV, int ldkv, int voff,
                                   int Lk, __bf16* __restrict__ VT) {
  GS_LOOP(idx, (size_t)B_ * Lk * C_) {
    size_t d = idx % C_, r = idx / C_;
    size_t k = r % Lk, b = r / Lk;
    VT[(b * C_ + d) * (size_t)Lk + k] = KV[(b * Lk + k) * (size_t)ldkv + voff + d];
  }
}
__global__ void broadcast_rep_kernel(const float* __restrict__ gfv, __bf16* __restrict__ out) {
  GS_LOOP(idx, (size_t)B_ * N_ * C_)
    out[idx] = f2bf(gfv[(idx / ((size_t)N_ * C_)) * C_ + (idx % C_)]);
}
__global__ void make_pairs_kernel(const float* __restrict__ feat,
                                  __bf16* __restrict__ fe, __bf16* __restrict__ fs) {
  GS_LOOP(idx, (size_t)B_ * L2_ * C_) {
    size_t c = idx % C_, r = idx / C_;
    size_t ij = r % L2_, b = r / L2_;
    size_t i = ij / N_, j = ij % N_;
    fe[idx] = f2bf(feat[(b * N_ + i) * (size_t)C_ + c]);   // repeat
    fs[idx] = f2bf(feat[(b * N_ + j) * (size_t)C_ + c]);   // tile
  }
}
__global__ void msg_build_kernel(const float* __restrict__ Vix, const float* __restrict__ Vjx,
                                 const float* __restrict__ elin, float* __restrict__ t) {
  GS_LOOP(idx, (size_t)B_ * L2_ * C_) {
    size_t c = idx % C_, r = idx / C_;
    size_t ij = r % L2_, b = r / L2_;
    size_t i = ij / N_, j = ij % N_;
    t[idx] = Vix[(b * N_ + i) * (size_t)C_ + c] + Vjx[(b * N_ + j) * (size_t)C_ + c] + elin[idx];
  }
}
__global__ void edge_update_kernel(const float* __restrict__ eo, const float* __restrict__ t,
                                   const float* __restrict__ mean, const float* __restrict__ inv,
                                   const float* __restrict__ g, const float* __restrict__ bt,
                                   float* __restrict__ enf, __bf16* __restrict__ enb) {
  GS_LOOP(idx, (size_t)B_ * L2_ * C_) {
    size_t ch = (idx / C_) % L2_;
    float v = eo[idx] + fmaxf(g[ch] * (t[idx] - mean[ch]) * inv[ch] + bt[ch], 0.f);
    enf[idx] = v; enb[idx] = f2bf(v);
  }
}
__global__ void edge_softmax_agg_kernel(const float* __restrict__ edge,
                                        const float* __restrict__ Ujx,
                                        float* __restrict__ agg) {
  GS_LOOP(idx, (size_t)B_ * N_ * C_) {
    size_t c = idx % C_, bi = idx / C_;
    size_t i = bi % N_, b = bi / N_;
    size_t base = ((b * N_ + i) * (size_t)N_) * C_ + c;
    float ss[N_]; float mx = -3.4e38f;
#pragma unroll
    for (int j = 0; j < N_; ++j) {
      float s = 1.f / (1.f + __expf(-edge[base + (size_t)j * C_]));
      ss[j] = s; mx = fmaxf(mx, s);
    }
    float sum = 0.f;
#pragma unroll
    for (int j = 0; j < N_; ++j) { ss[j] = __expf(ss[j] - mx); sum += ss[j]; }
    float a = 0.f;
    size_t ub = b * (size_t)(N_ * C_) + c;
#pragma unroll
    for (int j = 0; j < N_; ++j) a += ss[j] * Ujx[ub + (size_t)j * C_];
    agg[bi * C_ + c] = a / sum;
  }
}
__global__ void x_update_pre_kernel(const float* __restrict__ xU, const float* __restrict__ agg,
                                    float* __restrict__ xn) {
  GS_LOOP(idx, (size_t)B_ * N_ * C_) xn[idx] = xU[idx] + agg[idx] * (1.f / N_);
}
__global__ void bn_apply_x_kernel(const float* __restrict__ res, const float* __restrict__ xn,
                                  const float* __restrict__ mean, const float* __restrict__ inv,
                                  const float* __restrict__ g, const float* __restrict__ bt,
                                  float* __restrict__ outf, __bf16* __restrict__ outb) {
  GS_LOOP(idx, (size_t)B_ * N_ * C_) {
    size_t n = (idx / C_) % N_;
    float v = res[idx] + fmaxf(g[n] * (xn[idx] - mean[n]) * inv[n] + bt[n], 0.f);
    outf[idx] = v; outb[idx] = f2bf(v);
  }
}
__global__ void add2_kernel(const float* __restrict__ a, const float* __restrict__ b,
                            float* __restrict__ yf, __bf16* __restrict__ yb, size_t total) {
  GS_LOOP(idx, total) {
    float v = a[idx] + b[idx];
    yf[idx] = v;
    if (yb) yb[idx] = f2bf(v);
  }
}

// ---------------------------------------------------------------------------
static inline int ew_blocks(size_t total) {
  size_t b = (total + 255) / 256;
  return (int)(b > 1048576 ? 1048576 : b);
}
static inline void gemm(const __bf16* A, const __bf16* W, const float* bias,
                        float* Cf, __bf16* Cb,
                        int M, int N, int K, int lda, int ldw, int ldc, int relu,
                        hipStream_t s) {
  long long tiles = (long long)(M / 32) * (N / 64);
  gemm_nt_kernel<<<(int)((tiles + 7) / 8), 256, 0, s>>>(A, W, bias, Cf, Cb, M, N, K, lda, ldw, ldc, relu);
}

extern "C" void kernel_launch(void* const* d_in, const int* in_sizes, int n_in,
                              void* d_out, int out_size, void* d_ws, size_t ws_size,
                              hipStream_t stream) {
  (void)in_sizes; (void)n_in; (void)out_size; (void)ws_size;
  const float* node   = (const float*)d_in[0];
  const float* g1w    = (const float*)d_in[1];
  const float* g1b    = (const float*)d_in[2];
  const float* g1g    = (const float*)d_in[3];
  const float* g1bt   = (const float*)d_in[4];
  const float* g2w    = (const float*)d_in[5];
  const float* g2b    = (const float*)d_in[6];
  const float* g2g    = (const float*)d_in[7];
  const float* g2bt   = (const float*)d_in[8];
  const float* fam_wi = (const float*)d_in[9];
  const float* fam_bi = (const float*)d_in[10];
  const float* fam_wo = (const float*)d_in[11];
  const float* fam_bo = (const float*)d_in[12];
  const float* arm_wi = (const float*)d_in[13];
  const float* arm_bi = (const float*)d_in[14];
  const float* arm_wo = (const float*)d_in[15];
  const float* arm_bo = (const float*)d_in[16];
  const float* gem_w1 = (const float*)d_in[17];
  const float* gem_b1 = (const float*)d_in[18];
  const float* gem_w2 = (const float*)d_in[19];
  const float* gem_b2 = (const float*)d_in[20];
  const float* gem_lng= (const float*)d_in[21];
  const float* gem_lnb= (const float*)d_in[22];
  const float* natt_wi= (const float*)d_in[23];
  const float* natt_bi= (const float*)d_in[24];
  const float* natt_wo= (const float*)d_in[25];
  const float* natt_bo= (const float*)d_in[26];
  const float* gU     = (const float*)d_in[27];
  const float* gV     = (const float*)d_in[28];
  const float* gAw    = (const float*)d_in[29];
  const float* gBw    = (const float*)d_in[30];
  const float* gE     = (const float*)d_in[31];
  const float* bnv_g  = (const float*)d_in[32];
  const float* bnv_b  = (const float*)d_in[33];
  const float* bne_g  = (const float*)d_in[34];
  const float* bne_b  = (const float*)d_in[35];
  const float* fn_w1  = (const float*)d_in[36];
  const float* fn_b1  = (const float*)d_in[37];
  const float* fn_w2  = (const float*)d_in[38];
  const float* fn_b2  = (const float*)d_in[39];
  const float* fn_lng = (const float*)d_in[40];
  const float* fn_lnb = (const float*)d_in[41];

  // ---- byte bump allocator over d_ws ----
  char* base = (char*)d_ws;
  size_t off = 0;
  auto AF = [&](size_t n) { float*  p = (float*)(base + off);  off = (off + n*4 + 255) & ~(size_t)255; return p; };
  auto AB = [&](size_t n) { __bf16* p = (__bf16*)(base + off); off = (off + n*2 + 255) & ~(size_t)255; return p; };
  const size_t SML = (size_t)B_ * N_ * C_;        // 0.5M
  const size_t BIG = (size_t)B_ * L2_ * C_;       // 16M
  const size_t CC  = (size_t)C_ * C_;

  // bf16 weight mirrors
  __bf16* g1wB   = AB((size_t)4*C_*C_*N_);
  __bf16* g2wB   = AB(4*CC);
  __bf16* famwiB = AB(3*CC); __bf16* famwoB = AB(CC);
  __bf16* armwiB = AB(3*CC); __bf16* armwoB = AB(CC);
  __bf16* gemw1B = AB(4*CC); __bf16* gemw2B = AB(4*CC);
  __bf16* natwiB = AB(3*CC); __bf16* natwoB = AB(CC);
  __bf16* gUB = AB(3*CC); __bf16* gVB = AB(3*CC); __bf16* gAB = AB(3*CC);
  __bf16* gBB = AB(3*CC); __bf16* gEB = AB(3*CC);
  __bf16* fnw1B = AB(4*CC); __bf16* fnw2B = AB(4*CC);
  // small activations
  __bf16* nodeB = AB(SML);
  __bf16* xtB   = AB(SML);
  float*  gfa   = AF((size_t)B_*4*C_);
  __bf16* gfrB  = AB((size_t)B_*4*C_);
  float*  gf2   = AF((size_t)B_*C_);
  float*  gfv   = AF((size_t)B_*C_);
  __bf16* gfrepB= AB(SML);
  __bf16* qpB   = AB(SML);
  __bf16* kvpB  = AB(SML*2);
  __bf16* vt1B  = AB(SML);
  float*  s1    = AF((size_t)B_*H_*N_*N_);
  __bf16* p1B   = AB((size_t)B_*H_*N_*N_);
  __bf16* o1B   = AB(SML);
  float*  feat  = AF(SML);
  __bf16* qkvB  = AB(SML*3);
  __bf16* vtnB  = AB(SML);
  float*  tsm   = AF(SML);
  float*  nattF = AF(SML); __bf16* nattB = AB(SML);
  float*  xAF = AF(SML); __bf16* xAB = AB(SML);
  float*  xBF = AF(SML); __bf16* xBB = AB(SML);
  float*  Vix = AF(SML); float* Vjx = AF(SML);
  float*  Ujx = AF(SML); float* xU = AF(SML);
  float*  agg = AF(SML); float* xn = AF(SML);
  float*  noutF = AF(SML); __bf16* noutB = AB(SML);
  __bf16* h2B = AB(SML*4);
  float*  meanE = AF(L2_); float* invE = AF(L2_);
  float*  meanV = AF(N_);  float* invV = AF(N_);
  // big persistent
  __bf16* VT2b   = AB(BIG);
  __bf16* O2b    = AB(BIG);
  float*  feat2F = AF(BIG); __bf16* feat2B = AB(BIG);
  float*  tffn   = AF(BIG);
  float*  edgeAF = AF(BIG); __bf16* edgeABb = AB(BIG);
  float*  edgeBF = AF(BIG); __bf16* edgeBBb = AB(BIG);
  float*  elin   = AF(BIG);
  float*  tmsg   = AF(BIG);
  // 192MB overlay region
  char* Rg = base + off; off += (size_t)192 << 20;
  __bf16* QP2b  = (__bf16*)(Rg);                     // 32MB, phase A+B
  __bf16* KVP2b = (__bf16*)(Rg + ((size_t)32 << 20)); // 64MB, phase A+B
  __bf16* FSb   = (__bf16*)(Rg + ((size_t)96 << 20)); // 32MB, phase A
  __bf16* FEb   = (__bf16*)(Rg + ((size_t)128 << 20));// 32MB, phase A
  float*  S2    = (float*)(Rg + ((size_t)96 << 20));  // 64MB, phase B (16 bh)
  __bf16* P2b   = (__bf16*)(Rg + ((size_t)160 << 20));// 32MB, phase B
  __bf16* HBb   = (__bf16*)(Rg);                      // 128MB, phase C

  // ---- 0. weight/activation bf16 mirrors ----
  cvt_bf16_kernel<<<ew_blocks((size_t)4*C_*C_*N_), 256, 0, stream>>>(g1w, g1wB, (size_t)4*C_*C_*N_);
  cvt_bf16_kernel<<<ew_blocks(4*CC), 256, 0, stream>>>(g2w, g2wB, 4*CC);
  cvt_bf16_kernel<<<ew_blocks(3*CC), 256, 0, stream>>>(fam_wi, famwiB, 3*CC);
  cvt_bf16_kernel<<<ew_blocks(CC),   256, 0, stream>>>(fam_wo, famwoB, CC);
  cvt_bf16_kernel<<<ew_blocks(3*CC), 256, 0, stream>>>(arm_wi, armwiB, 3*CC);
  cvt_bf16_kernel<<<ew_blocks(CC),   256, 0, stream>>>(arm_wo, armwoB, CC);
  cvt_bf16_kernel<<<ew_blocks(4*CC), 256, 0, stream>>>(gem_w1, gemw1B, 4*CC);
  cvt_bf16_kernel<<<ew_blocks(4*CC), 256, 0, stream>>>(gem_w2, gemw2B, 4*CC);
  cvt_bf16_kernel<<<ew_blocks(3*CC), 256, 0, stream>>>(natt_wi, natwiB, 3*CC);
  cvt_bf16_kernel<<<ew_blocks(CC),   256, 0, stream>>>(natt_wo, natwoB, CC);
  cvt_bf16_kernel<<<ew_blocks(3*CC), 256, 0, stream>>>(gU, gUB, 3*CC);
  cvt_bf16_kernel<<<ew_blocks(3*CC), 256, 0, stream>>>(gV, gVB, 3*CC);
  cvt_bf16_kernel<<<ew_blocks(3*CC), 256, 0, stream>>>(gAw, gAB, 3*CC);
  cvt_bf16_kernel<<<ew_blocks(3*CC), 256, 0, stream>>>(gBw, gBB, 3*CC);
  cvt_bf16_kernel<<<ew_blocks(3*CC), 256, 0, stream>>>(gE, gEB, 3*CC);
  cvt_bf16_kernel<<<ew_blocks(4*CC), 256, 0, stream>>>(fn_w1, fnw1B, 4*CC);
  cvt_bf16_kernel<<<ew_blocks(4*CC), 256, 0, stream>>>(fn_w2, fnw2B, 4*CC);
  cvt_bf16_kernel<<<ew_blocks(SML),  256, 0, stream>>>(node, nodeB, SML);

  // ---- 1. global-feature branch ----
  transpose_node_kernel<<<ew_blocks(SML), 256, 0, stream>>>(node, xtB);
  gemm(xtB, g1wB, g1b, gfa, nullptr, B_, 4*C_, C_*N_, C_*N_, C_*N_, 4*C_, 0, stream);
  groupnorm_relu_kernel<<<B_*32, 32, 0, stream>>>(gfa, g1g, g1bt, nullptr, gfrB, 4*C_, 32);
  gemm(gfrB, g2wB, g2b, gf2, nullptr, B_, C_, 4*C_, 4*C_, 4*C_, C_, 0, stream);
  groupnorm_relu_kernel<<<B_*32, 32, 0, stream>>>(gf2, g2g, g2bt, gfv, nullptr, C_, 32);
  broadcast_rep_kernel<<<ew_blocks(SML), 256, 0, stream>>>(gfv, gfrepB);

  // ---- 2. fam MHA (q=node, kv=gf_rep), L=32 ----
  gemm(nodeB,  famwiB,      fam_bi,      nullptr, qpB,  B_*N_, C_,   C_, C_, C_, C_,   0, stream);
  gemm(gfrepB, famwiB + CC, fam_bi + C_, nullptr, kvpB, B_*N_, 2*C_, C_, C_, C_, 2*C_, 0, stream);
  transpose_v_kernel<<<ew_blocks(SML), 256, 0, stream>>>(kvpB, 2*C_, C_, N_, vt1B);
  attn_scores_kernel<<<dim3(1, B_*H_), 256, 0, stream>>>(qpB, C_, kvpB, 2*C_, s1, N_, N_, 0.125f, 0);
  softmax_rows_kernel<<<B_*H_*N_, 128, 0, stream>>>(s1, p1B, N_);
  attn_av_kernel<<<dim3(1, B_*H_), 256, 0, stream>>>(p1B, vt1B, o1B, C_, N_, N_, 0);
  gemm(o1B, famwoB, fam_bo, feat, nullptr, B_*N_, C_, C_, C_, C_, C_, 0, stream);

  // ---- 3. arm MHA over L2=1024 (16 chunks of 16 heads) ----
  make_pairs_kernel<<<ew_blocks(BIG), 256, 0, stream>>>(feat, FEb, FSb);
  gemm(FSb, armwiB,      arm_bi,      nullptr, QP2b,  B_*L2_, C_,   C_, C_, C_, C_,   0, stream);
  gemm(FEb, armwiB + CC, arm_bi + C_, nullptr, KVP2b, B_*L2_, 2*C_, C_, C_, C_, 2*C_, 0, stream);
  transpose_v_kernel<<<ew_blocks(BIG), 256, 0, stream>>>(KVP2b, 2*C_, C_, L2_, VT2b);
  for (int ck = 0; ck < 16; ++ck) {
    int bh0 = ck * 16;
    long long st = (long long)(L2_/16) * (L2_/32);
    attn_scores_kernel<<<dim3((int)((st+7)/8), 16), 256, 0, stream>>>(QP2b, C_, KVP2b, 2*C_, S2, L2_, L2_, 0.125f, bh0);
    softmax_rows_kernel<<<16 * L2_, 128, 0, stream>>>(S2, P2b, L2_);
    long long at = (long long)(L2_/16) * (HD_/32);
    attn_av_kernel<<<dim3((int)((at+7)/8), 16), 256, 0, stream>>>(P2b, VT2b, O2b, C_, L2_, L2_, bh0);
  }
  gemm(O2b, armwoB, arm_bo, feat2F, feat2B, B_*L2_, C_, C_, C_, C_, C_, 0, stream);

  // ---- 4. edge FFN + LN -> edge ----
  gemm(feat2B, gemw1B, gem_b1, nullptr, HBb, B_*L2_, 4*C_, C_, C_, C_, 4*C_, 1, stream);
  gemm(HBb, gemw2B, gem_b2, tffn, nullptr, B_*L2_, C_, 4*C_, 4*C_, 4*C_, C_, 0, stream);
  layernorm_kernel<<<B_*L2_, 256, 0, stream>>>(tffn, feat2F, gem_lng, gem_lnb, nullptr, edgeAF, edgeABb, C_);

  // ---- 5. natt self-attention on node ----
  gemm(nodeB, natwiB, natt_bi, nullptr, qkvB, B_*N_, 3*C_, C_, C_, C_, 3*C_, 0, stream);
  transpose_v_kernel<<<ew_blocks(SML), 256, 0, stream>>>(qkvB, 3*C_, 2*C_, N_, vtnB);
  attn_scores_kernel<<<dim3(1, B_*H_), 256, 0, stream>>>(qkvB, 3*C_, qkvB + C_, 3*C_, s1, N_, N_, 0.125f, 0);
  softmax_rows_kernel<<<B_*H_*N_, 128, 0, stream>>>(s1, p1B, N_);
  attn_av_kernel<<<dim3(1, B_*H_), 256, 0, stream>>>(p1B, vtnB, o1B, C_, N_, N_, 0);
  gemm(o1B, natwoB, natt_bo, tsm, nullptr, B_*N_, C_, C_, C_, C_, C_, 0, stream);
  add2_kernel<<<ew_blocks(SML), 256, 0, stream>>>(node, tsm, nattF, nattB, SML);

  // ---- 6. three gated-GNN layers ----
  float* xcf = nattF; __bf16* xcb = nattB;
  float* xnfF = xAF;  __bf16* xnfB = xAB;
  float* ecf = edgeAF; __bf16* ecb = edgeABb;
  float* enf = edgeBF; __bf16* enb = edgeBBb;
  for (int i = 0; i < 3; ++i) {
    const size_t WO = (size_t)i * CC;
    gemm(xcb, gAB + WO, nullptr, Vix,  nullptr, B_*N_,  C_, C_, C_, C_, C_, 0, stream);
    gemm(xcb, gBB + WO, nullptr, Vjx,  nullptr, B_*N_,  C_, C_, C_, C_, C_, 0, stream);
    gemm(ecb, gEB + WO, nullptr, elin, nullptr, B_*L2_, C_, C_, C_, C_, C_, 0, stream);
    msg_build_kernel<<<ew_blocks(BIG), 256, 0, stream>>>(Vix, Vjx, elin, tmsg);
    bn_stats_kernel<<<L2_, 256, 0, stream>>>(tmsg, B_, L2_, C_, meanE, invE);
    edge_update_kernel<<<ew_blocks(BIG), 256, 0, stream>>>(
        ecf, tmsg, meanE, invE, bne_g + (size_t)i*L2_, bne_b + (size_t)i*L2_, enf, enb);
    { float* tf = ecf; ecf = enf; enf = tf; __bf16* tb = ecb; ecb = enb; enb = tb; }
    gemm(xcb, gVB + WO, nullptr, Ujx, nullptr, B_*N_, C_, C_, C_, C_, C_, 0, stream);
    edge_softmax_agg_kernel<<<ew_blocks(SML), 256, 0, stream>>>(ecf, Ujx, agg);
    gemm(xcb, gUB + WO, nullptr, xU, nullptr, B_*N_, C_, C_, C_, C_, C_, 0, stream);
    x_update_pre_kernel<<<ew_blocks(SML), 256, 0, stream>>>(xU, agg, xn);
    bn_stats_kernel<<<N_, 256, 0, stream>>>(xn, B_, N_, C_, meanV, invV);
    bn_apply_x_kernel<<<ew_blocks(SML), 256, 0, stream>>>(
        xcf, xn, meanV, invV, bnv_g + (size_t)i*N_, bnv_b + (size_t)i*N_, xnfF, xnfB);
    xcf = xnfF; xcb = xnfB;
    if (xcf == xAF) { xnfF = xBF; xnfB = xBB; } else { xnfF = xAF; xnfB = xAB; }
  }
  add2_kernel<<<ew_blocks(SML), 256, 0, stream>>>(nattF, xcf, noutF, noutB, SML);

  // ---- 7. final FFN + LN + residual(node) -> d_out ----
  gemm(noutB, fnw1B, fn_b1, nullptr, h2B, B_*N_, 4*C_, C_, C_, C_, 4*C_, 1, stream);
  gemm(h2B, fnw2B, fn_b2, tsm, nullptr, B_*N_, C_, 4*C_, 4*C_, 4*C_, C_, 0, stream);
  layernorm_kernel<<<B_*N_, 256, 0, stream>>>(tsm, noutF, fn_lng, fn_lnb, node, (float*)d_out, nullptr, C_);
}
